// GPTSelfAttention_27247272526002
// MI455X (gfx1250) — matrix operand order repaired and checked by reading, not compiled, not run on hardware
//
#include <hip/hip_runtime.h>

// ---------------------------------------------------------------------------
// GPT self-attention forward for MI455X (gfx1250, wave32, WMMA).
// B=4, S=2048, D=1024, H=16, HD=64.  ~137 GFLOP, compute-bound -> bf16 WMMA.
// GEMMs: double-buffered LDS; A tiles via global_load_async_to_lds_b128,
// B (weight) tiles via the Tensor Data Mover (tensor_load_to_lds, TENSORcnt).
// ---------------------------------------------------------------------------

typedef __attribute__((ext_vector_type(16))) __bf16 v16bf;
typedef __attribute__((ext_vector_type(8)))  __bf16 v8bf;
typedef __attribute__((ext_vector_type(4)))  __bf16 v4bf;
typedef __attribute__((ext_vector_type(8)))  float  v8f;
typedef __attribute__((ext_vector_type(4)))  unsigned u32x4;
typedef __attribute__((ext_vector_type(8)))  int      i32x8;
typedef __attribute__((ext_vector_type(4)))  int      i32x4;

#define Bsz 4
#define Ssz 2048
#define Dsz 1024
#define Hsz 16
#define HD  64

// ---- helpers --------------------------------------------------------------

__device__ __forceinline__ __bf16 f2bf(float f) {
  unsigned u = __builtin_bit_cast(unsigned, f);
  u += 0x7fffu + ((u >> 16) & 1u);               // round-to-nearest-even
  unsigned short h = (unsigned short)(u >> 16);
  return __builtin_bit_cast(__bf16, h);
}

__device__ __forceinline__ unsigned ldsoff(const void* p) {
  // low 32 bits of a generic pointer into LDS == ds-space byte address
  return (unsigned)(unsigned long long)p;
}

// async global -> LDS, 16 bytes per active lane (ASYNCcnt path)
__device__ __forceinline__ void async_b128(unsigned lds, const void* gaddr) {
  asm volatile("global_load_async_to_lds_b128 %0, %1, off"
               :: "v"(lds), "v"(gaddr) : "memory");
}
template <int N>
__device__ __forceinline__ void wait_async_le() {
  asm volatile("s_wait_asynccnt %0" :: "n"(N) : "memory");
}

// Tensor Data Mover: 2D tile load into LDS. Dims/strides in elements (2B).
// D# packing per CDNA5 ISA 8.3/8.4 (group0: count/lds/global/type,
// group1: data_size + tensor_dim0/1 + tile_dim0/1 + dim0 stride).
// This toolchain exposes the 6-arg builtin; groups 2/3 and the extension
// operand are unused for 2D tiles -> zeros.
__device__ __forceinline__ void tdm_load_2d(unsigned lds_addr, const void* g,
                                            unsigned tensor_d0, unsigned tensor_d1,
                                            unsigned tile_d0, unsigned tile_d1,
                                            unsigned stride0) {
  unsigned long long ga = (unsigned long long)g;
  u32x4 g0;
  g0[0] = 1u;                                          // count=1, user D#
  g0[1] = lds_addr;                                    // LDS byte address
  g0[2] = (unsigned)(ga & 0xffffffffu);                // global_addr[31:0]
  g0[3] = (unsigned)((ga >> 32) & 0x01ffffffu)         // global_addr[56:32]
          | 0x80000000u;                               // type=2 ("image")
  i32x8 g1;
  g1[0] = (int)(1u << 16);                             // data_size=1 (2 bytes)
  g1[1] = (int)((tensor_d0 & 0xffffu) << 16);          // tensor_dim0[15:0]
  g1[2] = (int)((tensor_d0 >> 16) | ((tensor_d1 & 0xffffu) << 16));
  g1[3] = (int)((tensor_d1 >> 16) | (tile_d0 << 16));  // dim1 hi | tile_dim0
  g1[4] = (int)tile_d1;                                // tile_dim1, tile_dim2=0
  g1[5] = (int)stride0;                                // tensor_dim0_stride lo
  g1[6] = 0;                                           // stride0 hi | stride1 lo
  g1[7] = 0;
  i32x4 z4 = {0, 0, 0, 0};                             // groups 2/3 unused (2D)
  i32x8 z8 = {0, 0, 0, 0, 0, 0, 0, 0};
  __builtin_amdgcn_tensor_load_to_lds(g0, g1, z4, z4, z8, 0);
}

// A fragment: 16x32 bf16, row-major in LDS, element stride `ld`
__device__ __forceinline__ v16bf load_frag_a(const __bf16* base, int ld,
                                             int row0, int k0) {
  const int lane = threadIdx.x & 31;
  const int half = lane >> 4;
  const __bf16* p = base + (row0 + (lane & 15)) * ld + k0 + half * 8;
  v8bf lo = *(const v8bf*)p;          // K = k0 + half*8 .. +7
  v8bf hi = *(const v8bf*)(p + 16);   // K = k0 + 16 + half*8 .. +7
  return __builtin_shufflevector(lo, hi, 0,1,2,3,4,5,6,7,8,9,10,11,12,13,14,15);
}

// B fragment: 32x16 bf16, row-major [K][N] in LDS, stride `ld`; lane%16 = N column
__device__ __forceinline__ v16bf load_frag_b(const __bf16* base, int ld,
                                             int k0, int n0) {
  const int lane = threadIdx.x & 31;
  const int half = lane >> 4;
  const __bf16* p = base + (k0 + half * 8) * ld + n0 + (lane & 15);
  v16bf r;
#pragma unroll
  for (int s = 0; s < 8; ++s) { r[s] = p[s * ld]; r[8 + s] = p[(16 + s) * ld]; }
  return r;
}

__device__ __forceinline__ v8f wmma_bf16(v16bf a, v16bf b, v8f c) {
  return __builtin_amdgcn_wmma_f32_16x16x32_bf16(false, a, false, b,
                                                 (short)0, c, false, false);
}

// ---- kernel 1: fp32 -> bf16 downconvert -----------------------------------

__global__ __launch_bounds__(256) void cvt_bf16(const float4* __restrict__ in,
                                                v4bf* __restrict__ out, int n4) {
  int i = blockIdx.x * 256 + threadIdx.x;
  if (i >= n4) return;
  float4 f = in[i];
  v4bf o;
  o[0] = f2bf(f.x); o[1] = f2bf(f.y); o[2] = f2bf(f.z); o[3] = f2bf(f.w);
  out[i] = o;
}

// ---- kernel 2: fused QKV projection GEMM ----------------------------------
// X[8192,1024] (bf16) x W[1024,3072] (bf16) + bias -> scatter Q/K/V [B,H,S,64]
// 128x128 tile, BK=32, 8 waves (4 along M x 2 along N), 2x4 frags per wave.
// Double-buffered: A via async-to-LDS, B via TDM issued by wave 0.

__global__ __launch_bounds__(256) void qkv_gemm(
    const __bf16* __restrict__ X, const __bf16* __restrict__ W,
    const float* __restrict__ bias,
    __bf16* __restrict__ Qb, __bf16* __restrict__ Kb, __bf16* __restrict__ Vb) {
  __shared__ __bf16 As[2][128 * 32];
  __shared__ __bf16 Bs[2][32 * 128];

  const int tid   = threadIdx.x;
  const int wid   = tid >> 5;
  const int lane  = tid & 31;
  const int half  = lane >> 4;
  const int mBase = blockIdx.x * 128;
  const int nBase = blockIdx.y * 128;
  const int wm    = (wid & 3) * 32;   // wave row offset in tile
  const int wn    = (wid >> 2) * 64;  // wave col offset in tile

  v8f acc[2][4];
#pragma unroll
  for (int i = 0; i < 2; ++i)
#pragma unroll
    for (int j = 0; j < 4; ++j) acc[i][j] = (v8f)0.0f;

  auto stageA = [&](int buf, int k0) {
#pragma unroll
    for (int u = 0; u < 2; ++u) {                  // 512 x 16B chunks
      int c = tid * 2 + u;
      int row = c >> 2, col = (c & 3) * 8;
      async_b128(ldsoff(&As[buf][row * 32 + col]),
                 X + (mBase + row) * Dsz + k0 + col);
    }
  };
  auto stageB = [&](int buf, int k0) {             // one TDM per workgroup
    if (wid == 0)
      tdm_load_2d(ldsoff(&Bs[buf][0]), W + k0 * (3 * Dsz) + nBase,
                  3 * Dsz, Dsz, 128, 32, 3 * Dsz);
  };

  stageA(0, 0);
  stageB(0, 0);
  int cur = 0;
  for (int k0 = 0; k0 < Dsz; k0 += 32) {
    if (k0 + 32 < Dsz) {                           // prefetch next slab
      stageA(cur ^ 1, k0 + 32);
      stageB(cur ^ 1, k0 + 32);
      wait_async_le<2>();                          // oldest 2 (= cur tile) done
      __builtin_amdgcn_s_wait_tensorcnt(1);        // cur TDM done (wave 0)
    } else {
      wait_async_le<0>();
      __builtin_amdgcn_s_wait_tensorcnt(0);
    }
    __syncthreads();                               // cur tile visible to all

    v16bf a[2], b[4];
#pragma unroll
    for (int i = 0; i < 2; ++i) a[i] = load_frag_a(As[cur], 32, wm + i * 16, 0);
#pragma unroll
    for (int j = 0; j < 4; ++j) b[j] = load_frag_b(Bs[cur], 128, 0, wn + j * 16);
#pragma unroll
    for (int i = 0; i < 2; ++i)
#pragma unroll
      for (int j = 0; j < 4; ++j) acc[i][j] = wmma_bf16(a[i], b[j], acc[i][j]);

    __syncthreads();                               // cur free for reuse
    cur ^= 1;
  }

  // epilogue: + bias, scatter to Q (x 1/sqrt(64)), K, V in [B,H,S,HD]
#pragma unroll
  for (int i = 0; i < 2; ++i)
#pragma unroll
    for (int j = 0; j < 4; ++j) {
      int colg = nBase + wn + j * 16 + (lane & 15);  // [0,3072)
      int h = colg / (3 * HD);
      int r = colg % (3 * HD);
      float bv = bias[colg];
#pragma unroll
      for (int e = 0; e < 8; ++e) {
        int rowg = mBase + wm + i * 16 + e + 8 * half;  // m = b*S + s
        int bb = rowg >> 11, s = rowg & (Ssz - 1);
        float v = acc[i][j][e] + bv;
        __bf16* dst; int hd;
        if (r < HD)          { dst = Qb; hd = r;          v *= 0.125f; }
        else if (r < 2 * HD) { dst = Kb; hd = r - HD; }
        else                 { dst = Vb; hd = r - 2 * HD; }
        dst[(((bb * Hsz + h) * Ssz) + s) * HD + hd] = f2bf(v);
      }
    }
}

// ---- kernel 3: flash attention --------------------------------------------
// grid (S/64, H, B), 128 threads = 4 waves; wave w owns q rows [w*16, w*16+16)
// of a 64-row q tile and the full 128-wide kv tile -> softmax stats intra-wave.

__global__ __launch_bounds__(128) void attn(
    const __bf16* __restrict__ Qb, const __bf16* __restrict__ Kb,
    const __bf16* __restrict__ Vb, __bf16* __restrict__ ctx) {
  __shared__ __bf16 Qs[64 * 64];      //  8 KB
  __shared__ __bf16 Kt[64 * 128];     // 16 KB, K transposed: [d][kv]
  __shared__ __bf16 Vs[128 * 64];     // 16 KB, [kv][d]
  __shared__ __bf16 Ps[4][16 * 128];  // 16 KB, per-wave P tile

  const int tid  = threadIdx.x;
  const int wid  = tid >> 5;
  const int lane = tid & 31;
  const int half = lane >> 4;

  const int qBase = blockIdx.x * 64;
  const int h     = blockIdx.y;
  const int b     = blockIdx.z;
  const int bh    = (b * Hsz + h) * Ssz * HD;  // element offset of head slab

  // Q tile is contiguous [64][64] -> async copy (512 x 16B)
  const __bf16* Qg = Qb + bh + qBase * HD;
  for (int c = tid; c < 512; c += 128) async_b128(ldsoff(&Qs[c * 8]), Qg + c * 8);

  float run_max[8], run_sum[8];
  v8f o[4];
#pragma unroll
  for (int i = 0; i < 8; ++i) { run_max[i] = -1e30f; run_sum[i] = 0.0f; }
#pragma unroll
  for (int f = 0; f < 4; ++f) o[f] = (v8f)0.0f;

  const int tEnd = (qBase + 63) >> 7;
  for (int t = 0; t <= tEnd; ++t) {
    const int kvBase = t << 7;
    __syncthreads();  // previous tile's K/V reads done

    // V tile contiguous [128][64] -> async (1024 x 16B)
    const __bf16* Vg = Vb + bh + kvBase * HD;
    for (int c = tid; c < 1024; c += 128) async_b128(ldsoff(&Vs[c * 8]), Vg + c * 8);

    // K tile: thread tid transposes kv row `tid` into Kt[d][tid]
    const __bf16* Kg = Kb + bh + (kvBase + tid) * HD;
    v8bf kr[8];
#pragma unroll
    for (int u = 0; u < 8; ++u) kr[u] = *(const v8bf*)(Kg + u * 8);
#pragma unroll
    for (int u = 0; u < 8; ++u)
#pragma unroll
      for (int e = 0; e < 8; ++e) Kt[(u * 8 + e) * 128 + tid] = kr[u][e];

    wait_async_le<0>();
    __syncthreads();

    // S = Q * K^T  (scale already folded into Q)
    v8f sfr[8];
#pragma unroll
    for (int f = 0; f < 8; ++f) sfr[f] = (v8f)0.0f;
#pragma unroll
    for (int ks = 0; ks < HD; ks += 32) {
      v16bf a = load_frag_a(Qs, HD, wid * 16, ks);
#pragma unroll
      for (int f = 0; f < 8; ++f) {
        v16bf bb = load_frag_b(Kt, 128, ks, f * 16);
        sfr[f] = wmma_bf16(a, bb, sfr[f]);
      }
    }

    // causal mask (-10000 like reference) + per-row tile max
    float tmax[8];
#pragma unroll
    for (int i = 0; i < 8; ++i) tmax[i] = -1e30f;
#pragma unroll
    for (int f = 0; f < 8; ++f) {
      int kvg = kvBase + f * 16 + (lane & 15);
#pragma unroll
      for (int i = 0; i < 8; ++i) {
        int qg = qBase + wid * 16 + i + 8 * half;
        float s = sfr[f][i];
        if (kvg > qg) s = -10000.0f;
        sfr[f][i] = s;
        tmax[i] = fmaxf(tmax[i], s);
      }
    }
#pragma unroll
    for (int i = 0; i < 8; ++i)
      for (int x = 1; x <= 8; x <<= 1)
        tmax[i] = fmaxf(tmax[i], __shfl_xor(tmax[i], x, 32));  // 16-lane halves

    float alpha[8], tsum[8];
#pragma unroll
    for (int i = 0; i < 8; ++i) {
      float nm = fmaxf(run_max[i], tmax[i]);
      alpha[i] = __expf(run_max[i] - nm);
      run_max[i] = nm;
      tsum[i] = 0.0f;
    }

    // P = exp(S - max), stash bf16 into wave-private LDS (C-layout -> rows)
#pragma unroll
    for (int f = 0; f < 8; ++f)
#pragma unroll
      for (int i = 0; i < 8; ++i) {
        float p = __expf(sfr[f][i] - run_max[i]);
        tsum[i] += p;
        Ps[wid][(i + 8 * half) * 128 + f * 16 + (lane & 15)] = f2bf(p);
      }
#pragma unroll
    for (int i = 0; i < 8; ++i) {
      for (int x = 1; x <= 8; x <<= 1) tsum[i] += __shfl_xor(tsum[i], x, 32);
      run_sum[i] = run_sum[i] * alpha[i] + tsum[i];
    }
#pragma unroll
    for (int f = 0; f < 4; ++f)
#pragma unroll
      for (int i = 0; i < 8; ++i) o[f][i] *= alpha[i];

    __syncthreads();  // order Ps stores before A-fragment reloads

    // O += P * V
#pragma unroll
    for (int ks = 0; ks < 128; ks += 32) {
      v16bf a = load_frag_a(Ps[wid], 128, 0, ks);
#pragma unroll
      for (int f = 0; f < 4; ++f) {
        v16bf bb = load_frag_b(Vs, HD, ks, f * 16);
        o[f] = wmma_bf16(a, bb, o[f]);
      }
    }
  }

  // epilogue: normalize, merge heads into ctx [B,S,D] (bf16)
#pragma unroll
  for (int f = 0; f < 4; ++f)
#pragma unroll
    for (int i = 0; i < 8; ++i) {
      int qg = qBase + wid * 16 + i + 8 * half;
      float v = o[f][i] / run_sum[i];
      ctx[(b * Ssz + qg) * Dsz + h * HD + f * 16 + (lane & 15)] = f2bf(v);
    }
}

// ---- kernel 4: output dense GEMM ------------------------------------------
// ctx[8192,1024] (bf16) x Wd[1024,1024] (bf16) + bias -> out fp32
// Same double-buffered async + TDM pipeline as qkv_gemm.

__global__ __launch_bounds__(256) void dense_gemm(
    const __bf16* __restrict__ X, const __bf16* __restrict__ W,
    const float* __restrict__ bias, float* __restrict__ out) {
  __shared__ __bf16 As[2][128 * 32];
  __shared__ __bf16 Bs[2][32 * 128];

  const int tid   = threadIdx.x;
  const int wid   = tid >> 5;
  const int lane  = tid & 31;
  const int half  = lane >> 4;
  const int mBase = blockIdx.x * 128;
  const int nBase = blockIdx.y * 128;
  const int wm    = (wid & 3) * 32;
  const int wn    = (wid >> 2) * 64;

  v8f acc[2][4];
#pragma unroll
  for (int i = 0; i < 2; ++i)
#pragma unroll
    for (int j = 0; j < 4; ++j) acc[i][j] = (v8f)0.0f;

  auto stageA = [&](int buf, int k0) {
#pragma unroll
    for (int u = 0; u < 2; ++u) {
      int c = tid * 2 + u;
      int row = c >> 2, col = (c & 3) * 8;
      async_b128(ldsoff(&As[buf][row * 32 + col]),
                 X + (mBase + row) * Dsz + k0 + col);
    }
  };
  auto stageB = [&](int buf, int k0) {
    if (wid == 0)
      tdm_load_2d(ldsoff(&Bs[buf][0]), W + k0 * Dsz + nBase,
                  Dsz, Dsz, 128, 32, Dsz);
  };

  stageA(0, 0);
  stageB(0, 0);
  int cur = 0;
  for (int k0 = 0; k0 < Dsz; k0 += 32) {
    if (k0 + 32 < Dsz) {
      stageA(cur ^ 1, k0 + 32);
      stageB(cur ^ 1, k0 + 32);
      wait_async_le<2>();
      __builtin_amdgcn_s_wait_tensorcnt(1);
    } else {
      wait_async_le<0>();
      __builtin_amdgcn_s_wait_tensorcnt(0);
    }
    __syncthreads();

    v16bf a[2], b[4];
#pragma unroll
    for (int i = 0; i < 2; ++i) a[i] = load_frag_a(As[cur], 32, wm + i * 16, 0);
#pragma unroll
    for (int j = 0; j < 4; ++j) b[j] = load_frag_b(Bs[cur], 128, 0, wn + j * 16);
#pragma unroll
    for (int i = 0; i < 2; ++i)
#pragma unroll
      for (int j = 0; j < 4; ++j) acc[i][j] = wmma_bf16(a[i], b[j], acc[i][j]);

    __syncthreads();
    cur ^= 1;
  }

#pragma unroll
  for (int i = 0; i < 2; ++i)
#pragma unroll
    for (int j = 0; j < 4; ++j) {
      int colg = nBase + wn + j * 16 + (lane & 15);
      float bv = bias[colg];
#pragma unroll
      for (int e = 0; e < 8; ++e) {
        int rowg = mBase + wm + i * 16 + e + 8 * half;
        out[rowg * Dsz + colg] = acc[i][j][e] + bv;
      }
    }
}

// ---- launch ---------------------------------------------------------------

extern "C" void kernel_launch(void* const* d_in, const int* in_sizes, int n_in,
                              void* d_out, int out_size, void* d_ws, size_t ws_size,
                              hipStream_t stream) {
  (void)in_sizes; (void)n_in; (void)out_size; (void)ws_size;

  const float* x   = (const float*)d_in[0];   // [4,2048,1024]
  const float* wq  = (const float*)d_in[1];   // [1024,3072]
  const float* bq  = (const float*)d_in[2];   // [3072]
  const float* wd  = (const float*)d_in[3];   // [1024,1024]
  const float* bd  = (const float*)d_in[4];   // [1024]
  float* out       = (float*)d_out;           // [4,2048,1024]

  char* ws = (char*)d_ws;
  __bf16* xb    = (__bf16*)(ws);                       // 16 MiB
  __bf16* wqkvb = (__bf16*)(ws + 16777216);            //  6 MiB
  __bf16* wdb   = (__bf16*)(ws + 23068672);            //  2 MiB
  __bf16* Qb    = (__bf16*)(ws + 25165824);            // 16 MiB [B,H,S,64]
  __bf16* Kb    = (__bf16*)(ws + 41943040);            // 16 MiB
  __bf16* Vb    = (__bf16*)(ws + 58720256);            // 16 MiB
  __bf16* ctxb  = (__bf16*)(ws + 75497472);            // 16 MiB [B,S,D]

  const int nx  = Bsz * Ssz * Dsz;      // 8388608
  const int nwq = Dsz * 3 * Dsz;        // 3145728
  const int nwd = Dsz * Dsz;            // 1048576

  cvt_bf16<<<nx  / 4 / 256, 256, 0, stream>>>((const float4*)x,  (v4bf*)xb,    nx  / 4);
  cvt_bf16<<<nwq / 4 / 256, 256, 0, stream>>>((const float4*)wq, (v4bf*)wqkvb, nwq / 4);
  cvt_bf16<<<nwd / 4 / 256, 256, 0, stream>>>((const float4*)wd, (v4bf*)wdb,   nwd / 4);

  qkv_gemm<<<dim3(64, 24), 256, 0, stream>>>(xb, wqkvb, bq, Qb, Kb, Vb);
  attn<<<dim3(Ssz / 64, Hsz, Bsz), 128, 0, stream>>>(Qb, Kb, Vb, ctxb);
  dense_gemm<<<dim3(64, 8), 256, 0, stream>>>(ctxb, wdb, bd, out);
}
